// StructuralEncoder_82918638617231
// MI455X (gfx1250) — matrix operand order repaired and checked by reading, not compiled, not run on hardware
//
#include <hip/hip_runtime.h>
#include <hip/hip_bf16.h>

// ---------------- problem constants (match reference) ----------------
#define N_NODES 2048
#define DFEAT   128
#define DMODEL  256
#define NHEAD   8
#define DHEAD   32
#define NLAYER  2
#define NEDGE   65536
#define NET     8
#define MAXDIST 10
#define MAXDEG  512
#define DFF     1024
#define DOUTSZ  128

typedef __attribute__((ext_vector_type(16))) __bf16 v16bf;
typedef __attribute__((ext_vector_type(8)))  __bf16 v8bf;
typedef __attribute__((ext_vector_type(8)))  float  v8f;

enum { F_GELU = 1, F_RES = 2, F_OUTF = 4, F_OUTB = 8, F_OUTBT = 16 };

// Build a 16x32 bf16 A-fragment in contiguous K order (B loads use the same order):
// lane L: row = L&15, contiguous K-run [half*16 .. half*16+15]
__device__ __forceinline__ v16bf load_a_frag(const __bf16* __restrict__ rowp, int half) {
    v8bf lo = *reinterpret_cast<const v8bf*>(rowp + half * 16);
    v8bf hi = *reinterpret_cast<const v8bf*>(rowp + half * 16 + 8);
    v16bf a;
#pragma unroll
    for (int i = 0; i < 8; ++i) { a[i] = lo[i]; a[8 + i] = hi[i]; }
    return a;
}

// -------- generic WMMA GEMM, 16(M) x 64(N) per wave: A-frag reused over 4 B-frags ----
// A row-major bf16, BT = B^T row-major bf16 (so B columns are contiguous K runs).
template <int FLAGS>
__global__ __launch_bounds__(256) void k_gemm(
    const __bf16* __restrict__ A, const __bf16* __restrict__ BT,
    const float* __restrict__ bias, const float* __restrict__ residual,
    float* __restrict__ outF, __bf16* __restrict__ outB, __bf16* __restrict__ outBT,
    int M, int Nn, int K)
{
    const int lane = threadIdx.x & 31;
    const int wave = threadIdx.x >> 5;
    const int tiles_n = Nn >> 6;                    // 64-wide N tiles
    const int total = (M >> 4) * tiles_n;
    const int t = blockIdx.x * 8 + wave;            // wave-uniform
    if (t >= total) return;
    const int i0 = (t / tiles_n) << 4;
    const int j0 = (t % tiles_n) << 6;
    const int half = lane >> 4;
    const int rowA = i0 + (lane & 15);
    const int colB = j0 + (lane & 15);

    v8f c[4] = {};
    const __bf16* arow = A + (size_t)rowA * K;
    const __bf16* brow = BT + (size_t)colB * K + half * 16;
    const size_t bstep = (size_t)16 * K;            // 16 columns ahead in BT
    for (int kk = 0; kk < K; kk += 32) {
        v16bf a = load_a_frag(arow + kk, half);
#pragma unroll
        for (int jq = 0; jq < 4; ++jq) {
            v16bf b = *reinterpret_cast<const v16bf*>(brow + (size_t)jq * bstep + kk);
            c[jq] = __builtin_amdgcn_wmma_f32_16x16x32_bf16(false, a, false, b,
                                                            (short)0, c[jq], false, false);
        }
    }
#pragma unroll
    for (int jq = 0; jq < 4; ++jq) {
        const int col = j0 + jq * 16 + (lane & 15);
        const float bc = bias ? bias[col] : 0.0f;
#pragma unroll
        for (int r = 0; r < 8; ++r) {
            const int row = i0 + half * 8 + r;
            const size_t idx = (size_t)row * Nn + col;
            float v = c[jq][r] + bc;
            if (FLAGS & F_GELU) v = 0.5f * v * (1.0f + erff(v * 0.70710678118654752f));
            if (FLAGS & F_RES)  v += residual[idx];
            if (FLAGS & F_OUTF) outF[idx] = v;
            if (FLAGS & F_OUTB) outB[idx] = (__bf16)v;
            if (FLAGS & F_OUTBT) outBT[(size_t)col * M + row] = (__bf16)v;
        }
    }
}

// -------- attention scores, all heads per tile, SPD bias computed inline -------------
// One wave owns a 16x16 (i,j) tile for ALL 8 heads: 8 WMMAs, one pos read per element,
// pure write of S = Q K^T * scale + spd_table[bucket(pos)]. No bias pre-pass needed.
__global__ __launch_bounds__(256) void k_scores(
    const __bf16* __restrict__ Q, const __bf16* __restrict__ Kb,
    const float* __restrict__ pos, const float* __restrict__ spd,
    float* __restrict__ S, float scale)
{
    __shared__ float spd_s[(MAXDIST + 1) * NHEAD];
    if (threadIdx.x < (MAXDIST + 1) * NHEAD) spd_s[threadIdx.x] = spd[threadIdx.x];
    __syncthreads();

    const int lane = threadIdx.x & 31;
    const int t = blockIdx.x * 8 + (threadIdx.x >> 5);   // 128*128 tiles exactly
    const int jt = t & 127;
    const int it = t >> 7;
    const int half = lane >> 4;
    const int rowQ = (it << 4) + (lane & 15);
    const int rowK = (jt << 4) + (lane & 15);
    const __bf16* qrow = Q + (size_t)rowQ * DMODEL;
    const __bf16* krow = Kb + (size_t)rowK * DMODEL + half * 16;

    v8f acc[NHEAD] = {};
#pragma unroll
    for (int h = 0; h < NHEAD; ++h) {
        v16bf a = load_a_frag(qrow + h * DHEAD, half);
        v16bf b = *reinterpret_cast<const v16bf*>(krow + h * DHEAD);
        acc[h] = __builtin_amdgcn_wmma_f32_16x16x32_bf16(false, a, false, b, (short)0,
                                                         acc[h], false, false);
    }
    const int col = (jt << 4) + (lane & 15);
#pragma unroll
    for (int r = 0; r < 8; ++r) {
        const int row = (it << 4) + half * 8 + r;
        const size_t off = (size_t)row * N_NODES + col;
        float f = fminf(fmaxf(pos[off] * (float)MAXDIST + 0.5f, 0.0f), (float)MAXDIST);
        const float* sp = spd_s + (int)f * NHEAD;
#pragma unroll
        for (int h = 0; h < NHEAD; ++h)
            S[(size_t)h * N_NODES * N_NODES + off] = acc[h][r] * scale + sp[h];
    }
}

// ---------------- per-edge bias scatter directly onto the score planes ---------------
__global__ void k_bias_edge(const int* __restrict__ ei, const int* __restrict__ et,
                            const float* __restrict__ etab, float* __restrict__ S) {
    int e = blockIdx.x * 256 + threadIdx.x;       // E
    int s = ei[e], d2 = ei[NEDGE + e], ty = et[e];
    const float* row = etab + ty * NHEAD;
    size_t off = (size_t)s * N_NODES + d2;
#pragma unroll
    for (int h2 = 0; h2 < NHEAD; ++h2)
        atomicAdd(&S[(size_t)h2 * N_NODES * N_NODES + off], row[h2]);
}

// ---------------- softmax over rows of S -> bf16 probs (single global pass) ----------
__global__ __launch_bounds__(256) void k_softmax(const float* __restrict__ S,
                                                 __bf16* __restrict__ P)
{
    __shared__ float red[256];
    const int tid = threadIdx.x;
    const size_t base = (size_t)blockIdx.x * N_NODES;   // blockIdx = h*N + row
    float sv[8];
#pragma unroll
    for (int q = 0; q < 8; ++q) sv[q] = S[base + tid + q * 256];
    float m = sv[0];
#pragma unroll
    for (int q = 1; q < 8; ++q) m = fmaxf(m, sv[q]);
    red[tid] = m; __syncthreads();
    for (int s = 128; s > 0; s >>= 1) {
        if (tid < s) red[tid] = fmaxf(red[tid], red[tid + s]);
        __syncthreads();
    }
    m = red[0]; __syncthreads();
    float sum = 0.0f;
#pragma unroll
    for (int q = 0; q < 8; ++q) { sv[q] = __expf(sv[q] - m); sum += sv[q]; }
    red[tid] = sum; __syncthreads();
    for (int s = 128; s > 0; s >>= 1) {
        if (tid < s) red[tid] += red[tid + s];
        __syncthreads();
    }
    const float inv = 1.0f / red[0];
#pragma unroll
    for (int q = 0; q < 8; ++q) P[base + tid + q * 256] = (__bf16)(sv[q] * inv);
}

// ---------------- attn_out = P @ V  (V held transposed [D, N]) -----------------------
__global__ __launch_bounds__(256) void k_av(const __bf16* __restrict__ P,
                                            const __bf16* __restrict__ VT,
                                            __bf16* __restrict__ O)
{
    const int lane = threadIdx.x & 31;
    const int t = blockIdx.x * 8 + (threadIdx.x >> 5);  // 8*128*2 tiles exactly
    const int dt = t & 1;
    const int it = (t >> 1) & 127;
    const int h  = t >> 8;
    const int half = lane >> 4;
    const int rowP = (it << 4) + (lane & 15);
    const int colD = h * DHEAD + dt * 16 + (lane & 15);

    const __bf16* prow = P + ((size_t)h * N_NODES + rowP) * N_NODES;
    const __bf16* vrow = VT + (size_t)colD * N_NODES + half * 16;
    v8f c = {};
    for (int kk = 0; kk < N_NODES; kk += 32) {
        v16bf a = load_a_frag(prow + kk, half);
        v16bf b = *reinterpret_cast<const v16bf*>(vrow + kk);
        c = __builtin_amdgcn_wmma_f32_16x16x32_bf16(false, a, false, b, (short)0, c,
                                                    false, false);
    }
#pragma unroll
    for (int r = 0; r < 8; ++r) {
        const int row = (it << 4) + half * 8 + r;
        O[(size_t)row * DMODEL + colD] = (__bf16)c[r];
    }
}

// ---------------- small elementwise / setup kernels ----------------------------------
__global__ void k_zero_i32(int* __restrict__ p, int n) {
    int i = blockIdx.x * 256 + threadIdx.x;
    if (i < n) p[i] = 0;
}
__global__ void k_degrees(const int* __restrict__ ei, int* __restrict__ degout,
                          int* __restrict__ degin) {
    int e = blockIdx.x * 256 + threadIdx.x;
    atomicAdd(&degout[ei[e]], 1);          // src row
    atomicAdd(&degin[ei[NEDGE + e]], 1);   // dst row
}
__global__ void k_f32_to_bf16(const float* __restrict__ in, __bf16* __restrict__ out,
                              int n) {
    int i = blockIdx.x * 256 + threadIdx.x;
    if (i < n) out[i] = (__bf16)in[i];
}
__global__ void k_convT(const float* __restrict__ W, __bf16* __restrict__ WT,
                        int K, int Nn) {
    int tid = blockIdx.x * 256 + threadIdx.x;
    if (tid >= K * Nn) return;
    int k = tid / Nn, n = tid % Nn;
    WT[(size_t)n * K + k] = (__bf16)W[tid];
}
__global__ void k_centrality(float* __restrict__ h, const int* __restrict__ degin,
                             const int* __restrict__ degout,
                             const float* __restrict__ din_emb,
                             const float* __restrict__ dout_emb) {
    int tid = blockIdx.x * 256 + threadIdx.x;     // N*D
    int n = tid >> 8, d = tid & 255;
    int di = min(degin[n], MAXDEG), dq = min(degout[n], MAXDEG);
    h[tid] += din_emb[(size_t)di * DMODEL + d] + dout_emb[(size_t)dq * DMODEL + d];
}
__global__ __launch_bounds__(256) void k_layernorm(const float* __restrict__ h,
                                                   const float* __restrict__ g,
                                                   const float* __restrict__ b,
                                                   __bf16* __restrict__ out) {
    __shared__ float red[256];
    const int tid = threadIdx.x;
    const size_t base = (size_t)blockIdx.x * DMODEL;
    float x = h[base + tid];
    red[tid] = x; __syncthreads();
    for (int s = 128; s > 0; s >>= 1) { if (tid < s) red[tid] += red[tid + s]; __syncthreads(); }
    const float mu = red[0] * (1.0f / DMODEL); __syncthreads();
    float d = x - mu;
    red[tid] = d * d; __syncthreads();
    for (int s = 128; s > 0; s >>= 1) { if (tid < s) red[tid] += red[tid + s]; __syncthreads(); }
    const float inv = rsqrtf(red[0] * (1.0f / DMODEL) + 1e-5f);
    out[base + tid] = (__bf16)(d * inv * g[tid] + b[tid]);
}

// ---------------- host launch ---------------------------------------------------------
extern "C" void kernel_launch(void* const* d_in, const int* in_sizes, int n_in,
                              void* d_out, int out_size, void* d_ws, size_t ws_size,
                              hipStream_t stream) {
    const float* x        = (const float*)d_in[0];
    const int*   eidx     = (const int*)d_in[1];
    const int*   etypes   = (const int*)d_in[2];
    const float* pos      = (const float*)d_in[3];
    const float* W_emb    = (const float*)d_in[4];
    const float* b_emb    = (const float*)d_in[5];
    const float* Wq       = (const float*)d_in[6];
    const float* Wk       = (const float*)d_in[7];
    const float* Wv       = (const float*)d_in[8];
    const float* Wo       = (const float*)d_in[9];
    const float* bq       = (const float*)d_in[10];
    const float* bk       = (const float*)d_in[11];
    const float* bv       = (const float*)d_in[12];
    const float* bo       = (const float*)d_in[13];
    const float* spd_tab  = (const float*)d_in[14];
    const float* edge_tab = (const float*)d_in[15];
    const float* din_emb  = (const float*)d_in[16];
    const float* dout_emb = (const float*)d_in[17];
    const float* ln1_g    = (const float*)d_in[18];
    const float* ln1_b    = (const float*)d_in[19];
    const float* ln2_g    = (const float*)d_in[20];
    const float* ln2_b    = (const float*)d_in[21];
    const float* W1       = (const float*)d_in[22];
    const float* b1       = (const float*)d_in[23];
    const float* W2       = (const float*)d_in[24];
    const float* b2       = (const float*)d_in[25];
    const float* W_out    = (const float*)d_in[26];
    const float* b_out    = (const float*)d_in[27];
    float* out = (float*)d_out;

    // workspace carve-up
    char* wp = (char*)d_ws;
    auto carve = [&](size_t bytes) { void* p = wp; wp += (bytes + 255) & ~(size_t)255; return p; };
    float*  hbuf  = (float*)carve((size_t)N_NODES * DMODEL * 4);
    __bf16* xn    = (__bf16*)carve((size_t)N_NODES * DMODEL * 2);   // LN out / final h bf16
    __bf16* qb    = (__bf16*)carve((size_t)N_NODES * DMODEL * 2);
    __bf16* kb    = (__bf16*)carve((size_t)N_NODES * DMODEL * 2);
    __bf16* vTb   = (__bf16*)carve((size_t)DMODEL * N_NODES * 2);   // V transposed [D,N]
    __bf16* attnb = (__bf16*)carve((size_t)N_NODES * DMODEL * 2);
    __bf16* ffnb  = (__bf16*)carve((size_t)N_NODES * DFF * 2);
    __bf16* xb    = (__bf16*)carve((size_t)N_NODES * DFEAT * 2);
    int* degin    = (int*)carve(N_NODES * 4);
    int* degout   = (int*)carve(N_NODES * 4);
    __bf16* WembT = (__bf16*)carve((size_t)DFEAT * DMODEL * 2);
    __bf16* WqT[NLAYER]; __bf16* WkT[NLAYER]; __bf16* WvT[NLAYER]; __bf16* WoT[NLAYER];
    __bf16* W1T[NLAYER]; __bf16* W2T[NLAYER];
    for (int l = 0; l < NLAYER; ++l) {
        WqT[l] = (__bf16*)carve((size_t)DMODEL * DMODEL * 2);
        WkT[l] = (__bf16*)carve((size_t)DMODEL * DMODEL * 2);
        WvT[l] = (__bf16*)carve((size_t)DMODEL * DMODEL * 2);
        WoT[l] = (__bf16*)carve((size_t)DMODEL * DMODEL * 2);
        W1T[l] = (__bf16*)carve((size_t)DMODEL * DFF * 2);
        W2T[l] = (__bf16*)carve((size_t)DFF * DMODEL * 2);
    }
    __bf16* WoutT = (__bf16*)carve((size_t)DMODEL * DOUTSZ * 2);
    float*  Sbuf  = (float*)carve((size_t)NHEAD * N_NODES * N_NODES * 4);  // scores
    __bf16* Pbuf  = (__bf16*)carve((size_t)NHEAD * N_NODES * N_NODES * 2); // probs

    const float scale = 0.17677669529663687f;   // 1/sqrt(32)

    // --- degrees ---
    k_zero_i32<<<(N_NODES + 255) / 256, 256, 0, stream>>>(degin, N_NODES);
    k_zero_i32<<<(N_NODES + 255) / 256, 256, 0, stream>>>(degout, N_NODES);
    k_degrees<<<NEDGE / 256, 256, 0, stream>>>(eidx, degout, degin);

    // --- weight conversion (bf16, transposed) ---
    k_convT<<<(DFEAT * DMODEL + 255) / 256, 256, 0, stream>>>(W_emb, WembT, DFEAT, DMODEL);
    for (int l = 0; l < NLAYER; ++l) {
        const size_t wdd = (size_t)l * DMODEL * DMODEL;
        k_convT<<<(DMODEL * DMODEL + 255) / 256, 256, 0, stream>>>(Wq + wdd, WqT[l], DMODEL, DMODEL);
        k_convT<<<(DMODEL * DMODEL + 255) / 256, 256, 0, stream>>>(Wk + wdd, WkT[l], DMODEL, DMODEL);
        k_convT<<<(DMODEL * DMODEL + 255) / 256, 256, 0, stream>>>(Wv + wdd, WvT[l], DMODEL, DMODEL);
        k_convT<<<(DMODEL * DMODEL + 255) / 256, 256, 0, stream>>>(Wo + wdd, WoT[l], DMODEL, DMODEL);
        k_convT<<<(DMODEL * DFF + 255) / 256, 256, 0, stream>>>(W1 + (size_t)l * DMODEL * DFF, W1T[l], DMODEL, DFF);
        k_convT<<<(DFF * DMODEL + 255) / 256, 256, 0, stream>>>(W2 + (size_t)l * DFF * DMODEL, W2T[l], DFF, DMODEL);
    }
    k_convT<<<(DMODEL * DOUTSZ + 255) / 256, 256, 0, stream>>>(W_out, WoutT, DMODEL, DOUTSZ);

    // --- embedding: h = x @ W_emb + b_emb ---
    k_f32_to_bf16<<<(N_NODES * DFEAT) / 256, 256, 0, stream>>>(x, xb, N_NODES * DFEAT);
    {
        int tiles = (N_NODES / 16) * (DMODEL / 64);
        k_gemm<F_OUTF><<<(tiles + 7) / 8, 256, 0, stream>>>(
            xb, WembT, b_emb, nullptr, hbuf, nullptr, nullptr, N_NODES, DMODEL, DFEAT);
    }

    for (int l = 0; l < NLAYER; ++l) {
        // centrality encodings
        k_centrality<<<(N_NODES * DMODEL) / 256, 256, 0, stream>>>(
            hbuf, degin, degout,
            din_emb + (size_t)l * (MAXDEG + 1) * DMODEL,
            dout_emb + (size_t)l * (MAXDEG + 1) * DMODEL);
        // LN1
        k_layernorm<<<N_NODES, 256, 0, stream>>>(hbuf, ln1_g + l * DMODEL,
                                                 ln1_b + l * DMODEL, xn);
        // Q / K / V projections (V written transposed for the AV WMMA)
        int tiles = (N_NODES / 16) * (DMODEL / 64);
        k_gemm<F_OUTB><<<(tiles + 7) / 8, 256, 0, stream>>>(
            xn, WqT[l], bq + l * DMODEL, nullptr, nullptr, qb, nullptr, N_NODES, DMODEL, DMODEL);
        k_gemm<F_OUTB><<<(tiles + 7) / 8, 256, 0, stream>>>(
            xn, WkT[l], bk + l * DMODEL, nullptr, nullptr, kb, nullptr, N_NODES, DMODEL, DMODEL);
        k_gemm<F_OUTBT><<<(tiles + 7) / 8, 256, 0, stream>>>(
            xn, WvT[l], bv + l * DMODEL, nullptr, nullptr, nullptr, vTb, N_NODES, DMODEL, DMODEL);
        // scores for all heads with inline SPD bias (pure write), then edge scatter
        k_scores<<<(128 * 128) / 8, 256, 0, stream>>>(
            qb, kb, pos, spd_tab + (size_t)l * (MAXDIST + 1) * NHEAD, Sbuf, scale);
        k_bias_edge<<<NEDGE / 256, 256, 0, stream>>>(
            eidx, etypes, edge_tab + (size_t)l * NET * NHEAD, Sbuf);
        // softmax, P@V
        k_softmax<<<NHEAD * N_NODES, 256, 0, stream>>>(Sbuf, Pbuf);
        k_av<<<(NHEAD * 128 * 2) / 8, 256, 0, stream>>>(Pbuf, vTb, attnb);
        // h += attn_out @ Wo + bo
        k_gemm<F_RES | F_OUTF><<<(tiles + 7) / 8, 256, 0, stream>>>(
            attnb, WoT[l], bo + l * DMODEL, hbuf, hbuf, nullptr, nullptr, N_NODES, DMODEL, DMODEL);
        // FFN: h += GELU(LN2(h) @ W1 + b1) @ W2 + b2
        k_layernorm<<<N_NODES, 256, 0, stream>>>(hbuf, ln2_g + l * DMODEL,
                                                 ln2_b + l * DMODEL, xn);
        int tiles1 = (N_NODES / 16) * (DFF / 64);
        k_gemm<F_GELU | F_OUTB><<<(tiles1 + 7) / 8, 256, 0, stream>>>(
            xn, W1T[l], b1 + l * DFF, nullptr, nullptr, ffnb, nullptr, N_NODES, DFF, DMODEL);
        k_gemm<F_RES | F_OUTF><<<(tiles + 7) / 8, 256, 0, stream>>>(
            ffnb, W2T[l], b2 + l * DMODEL, hbuf, hbuf, nullptr, nullptr, N_NODES, DMODEL, DFF);
    }

    // output head: out = h @ W_out + b_out
    k_f32_to_bf16<<<(N_NODES * DMODEL) / 256, 256, 0, stream>>>(hbuf, xn, N_NODES * DMODEL);
    int tilesO = (N_NODES / 16) * (DOUTSZ / 64);
    k_gemm<F_OUTF><<<(tilesO + 7) / 8, 256, 0, stream>>>(
        xn, WoutT, b_out, nullptr, out, nullptr, nullptr, N_NODES, DOUTSZ, DMODEL);
}
